// RGAT_PyG_34754875359992
// MI455X (gfx1250) — matrix operand order repaired and checked by reading, not compiled, not run on hardware
//
#include <hip/hip_runtime.h>
#include <hip/hip_bf16.h>

#define NN   50000
#define EE   800000
#define IND  128
#define OUTD 64
#define RR   8
#define NEG_SLOPE 0.2f

typedef __attribute__((ext_vector_type(16))) __bf16    v16bf;
typedef __attribute__((ext_vector_type(8)))  float     v8f;
typedef __attribute__((ext_vector_type(8)))  unsigned  v8u;

// fp32 -> bf16 bits, round-to-nearest-even
__device__ __forceinline__ unsigned bfbits(float f) {
    unsigned u = __float_as_uint(f);
    return (u + 0x7FFFu + ((u >> 16) & 1u)) >> 16;
}
// pack two fp32 into one dword of two bf16 (f0 -> low half)
__device__ __forceinline__ unsigned packbf(float f0, float f1) {
    return (bfbits(f1) << 16) | (bfbits(f0) & 0xFFFFu);
}

__device__ __forceinline__ void atomAddF(float* p, float v) {
    unsafeAtomicAdd(p, v);   // native global_atomic_add_f32
}

// ---------------------------------------------------------------------------
// Kernel 0: convert features to packed bf16 pairs (row-major [N, 64] dwords)
// ---------------------------------------------------------------------------
__global__ void feat_to_bf16(const float* __restrict__ feat, unsigned* __restrict__ featbf) {
    const int i = blockIdx.x * blockDim.x + threadIdx.x;     // over N * IND/2
    if (i >= NN * (IND / 2)) return;
    const float2 f = ((const float2*)feat)[i];
    featbf[i] = packbf(f.x, f.y);
}

// ---------------------------------------------------------------------------
// Kernel 1: h[r,n,o] = sum_i features[n,i] * weight[r,i,o]   (bf16 WMMA, f32 acc)
// plus fused epilogue: qh[r,n] = h[r,n,:]·q,  kh[r,n] = h[r,n,:]·k
// One wave -> 16 rows x 64 cols of h. Block = 8 waves = 128 rows.
// A: pre-packed bf16 in global, 2x b128 per lane per K step, no conversion ALU.
// B: fragment-major packed-bf16 LDS cache, 2x ds_load_b128 per fragment.
// ---------------------------------------------------------------------------
__global__ __launch_bounds__(256) void rgemm_wmma(const unsigned* __restrict__ featbf,
                                                  const float* __restrict__ w,
                                                  const float* __restrict__ q,
                                                  const float* __restrict__ k,
                                                  float* __restrict__ h,
                                                  float* __restrict__ qh,
                                                  float* __restrict__ kh) {
    const int r      = blockIdx.y;
    const int waveId = threadIdx.x >> 5;
    const int lane   = threadIdx.x & 31;
    const int lo     = lane & 15;
    const int hi     = lane >> 4;

    // fragment-major weight cache: index ((kk*4+nt)*32 + lane) -> 8 dwords (16 bf16)
    __shared__ __align__(32) unsigned wlds[4096];   // 16 KB

    const float* wr = w + (size_t)r * IND * OUTD;
    for (int idx2 = threadIdx.x; idx2 < 4096; idx2 += 256) {
        const int i2  = idx2 & 7;          // bf16-pair within fragment
        const int ln  = (idx2 >> 3) & 31;  // destination lane
        const int nt  = (idx2 >> 8) & 3;   // N tile
        const int kk  = (idx2 >> 10) & 3;  // K step
        const int K   = kk * 32 + (i2 >> 2) * 16 + (ln >> 4) * 8 + 2 * (i2 & 3);
        const int Nc  = nt * 16 + (ln & 15);
        wlds[idx2] = packbf(wr[K * OUTD + Nc], wr[(K + 1) * OUTD + Nc]);
    }
    __syncthreads();

    const int mbase = (blockIdx.x * 8 + waveId) * 16;
    const int row   = mbase + lo;                         // A-matrix: lane -> row M
    const int rowc  = row < NN ? row : NN - 1;            // clamp; pad rows never stored
    const uint4* fb = (const uint4*)(featbf + (size_t)rowc * (IND / 2));

    v8f acc0 = {}, acc1 = {}, acc2 = {}, acc3 = {};
    const v8u* wfrag = (const v8u*)wlds;

#pragma unroll
    for (int kk = 0; kk < 4; ++kk) {
        // A fragment: elems 0..7 -> K = 32kk + hi*8 + i ; elems 8..15 -> +16
        const uint4 a0 = fb[kk * 4 + hi];
        const uint4 a1 = fb[kk * 4 + 2 + hi];
        v8u au;
        au[0] = a0.x; au[1] = a0.y; au[2] = a0.z; au[3] = a0.w;
        au[4] = a1.x; au[5] = a1.y; au[6] = a1.z; au[7] = a1.w;
        const v16bf a = __builtin_bit_cast(v16bf, au);

        // B fragments: one 32-byte LDS load each
        const v16bf b0v = __builtin_bit_cast(v16bf, wfrag[(kk * 4 + 0) * 32 + lane]);
        const v16bf b1v = __builtin_bit_cast(v16bf, wfrag[(kk * 4 + 1) * 32 + lane]);
        const v16bf b2v = __builtin_bit_cast(v16bf, wfrag[(kk * 4 + 2) * 32 + lane]);
        const v16bf b3v = __builtin_bit_cast(v16bf, wfrag[(kk * 4 + 3) * 32 + lane]);

        acc0 = __builtin_amdgcn_wmma_f32_16x16x32_bf16(false, a, false, b0v, (short)0, acc0, false, false);
        acc1 = __builtin_amdgcn_wmma_f32_16x16x32_bf16(false, a, false, b1v, (short)0, acc1, false, false);
        acc2 = __builtin_amdgcn_wmma_f32_16x16x32_bf16(false, a, false, b2v, (short)0, acc2, false, false);
        acc3 = __builtin_amdgcn_wmma_f32_16x16x32_bf16(false, a, false, b3v, (short)0, acc3, false, false);
    }

    // ---- store h tile.  C/D layout: VGPR j -> M = mbase + j + 8*hi, N = nt*16+lo
#pragma unroll
    for (int j = 0; j < 8; ++j) {
        const int m = mbase + j + 8 * hi;
        if (m < NN) {
            float* hrow = h + ((size_t)r * NN + m) * OUTD;
            hrow[ 0 + lo] = acc0[j];
            hrow[16 + lo] = acc1[j];
            hrow[32 + lo] = acc2[j];
            hrow[48 + lo] = acc3[j];
        }
    }

    // ---- fused q/k projection: qh[r,m] = sum_N h[m,N]*q[N]
    const float qc0 = q[lo], qc1 = q[16 + lo], qc2 = q[32 + lo], qc3 = q[48 + lo];
    const float kc0 = k[lo], kc1 = k[16 + lo], kc2 = k[32 + lo], kc3 = k[48 + lo];
#pragma unroll
    for (int j = 0; j < 8; ++j) {
        float sq = acc0[j] * qc0 + acc1[j] * qc1 + acc2[j] * qc2 + acc3[j] * qc3;
        float sk = acc0[j] * kc0 + acc1[j] * kc1 + acc2[j] * kc2 + acc3[j] * kc3;
#pragma unroll
        for (int off = 1; off <= 8; off <<= 1) {   // reduce over the 16-lane N group
            sq += __shfl_xor(sq, off, 32);
            sk += __shfl_xor(sk, off, 32);
        }
        if (lo == 0) {
            const int m = mbase + j + 8 * hi;
            if (m < NN) {
                qh[r * NN + m] = sq;
                kh[r * NN + m] = sk;
            }
        }
    }
}

// ---------------------------------------------------------------------------
// Kernel 2: init output/den/maxkey
// ---------------------------------------------------------------------------
__global__ void init_buffers(float* __restrict__ out, float* __restrict__ den,
                             unsigned* __restrict__ maxkey) {
    const int i = blockIdx.x * blockDim.x + threadIdx.x;
    if (i < NN * OUTD) out[i] = 0.0f;
    if (i < NN) { den[i] = 0.0f; maxkey[i] = 0u; }
}

// ordered-uint mapping of float for atomicMax
__device__ __forceinline__ unsigned fkey(float f) {
    unsigned u = __float_as_uint(f);
    return (u & 0x80000000u) ? ~u : (u | 0x80000000u);
}
__device__ __forceinline__ float unkey(unsigned key) {
    unsigned u = (key & 0x80000000u) ? (key & 0x7FFFFFFFu) : ~key;
    return __uint_as_float(u);
}

// ---------------------------------------------------------------------------
// Kernel 3: per-edge logit = leaky_relu(qh[et,dst] + kh[et,src]); segment max
// ---------------------------------------------------------------------------
__global__ void edge_logit_max(const int* __restrict__ adj, const int* __restrict__ et,
                               const float* __restrict__ qh, const float* __restrict__ kh,
                               float* __restrict__ exl, unsigned* __restrict__ maxkey) {
    const int e = blockIdx.x * blockDim.x + threadIdx.x;
    if (e >= EE) return;
    const int s = adj[e], d = adj[EE + e], r = et[e];
    float x = qh[r * NN + d] + kh[r * NN + s];
    x = (x > 0.0f) ? x : NEG_SLOPE * x;
    exl[e] = x;
    atomicMax(&maxkey[d], fkey(x));
}

// ---------------------------------------------------------------------------
// Kernel 4: ex = exp(logit - max[dst]); segment sum
// ---------------------------------------------------------------------------
__global__ void edge_exp_sum(const int* __restrict__ adj, float* __restrict__ exl,
                             const unsigned* __restrict__ maxkey, float* __restrict__ den) {
    const int e = blockIdx.x * blockDim.x + threadIdx.x;
    if (e >= EE) return;
    const int d = adj[EE + e];
    const float m = unkey(maxkey[d]);
    const float ex = __expf(exl[e] - m);
    exl[e] = ex;
    atomAddF(&den[d], ex);
}

// ---------------------------------------------------------------------------
// Kernel 5: scatter messages: aggr[dst,:] += (ex/den)*h[et,src,:]; wave per edge,
// each lane handles a coalesced float2 of the 64-wide message.
// ---------------------------------------------------------------------------
__global__ __launch_bounds__(256) void edge_scatter(const int* __restrict__ adj,
                                                    const int* __restrict__ et,
                                                    const float* __restrict__ h,
                                                    const float* __restrict__ exl,
                                                    const float* __restrict__ den,
                                                    float* __restrict__ aggr) {
    const long long t = (long long)blockIdx.x * blockDim.x + threadIdx.x;
    const int e    = (int)(t >> 5);
    const int lane = (int)(t & 31);
    if (e >= EE) return;
    const int s = adj[e], d = adj[EE + e], r = et[e];
    const float coeff = exl[e] / (den[d] + 1e-16f);
    const float2 hv = ((const float2*)(h + ((size_t)r * NN + s) * OUTD))[lane];
    float* dst = aggr + (size_t)d * OUTD + 2 * lane;
    atomAddF(dst + 0, coeff * hv.x);
    atomAddF(dst + 1, coeff * hv.y);
}

// ---------------------------------------------------------------------------
// Kernel 6: out = relu(aggr + bias)
// ---------------------------------------------------------------------------
__global__ void bias_relu(float* __restrict__ out, const float* __restrict__ bias) {
    const int i = blockIdx.x * blockDim.x + threadIdx.x;
    if (i >= NN * OUTD) return;
    const float v = out[i] + bias[i & (OUTD - 1)];
    out[i] = v > 0.0f ? v : 0.0f;
}

// ---------------------------------------------------------------------------
extern "C" void kernel_launch(void* const* d_in, const int* in_sizes, int n_in,
                              void* d_out, int out_size, void* d_ws, size_t ws_size,
                              hipStream_t stream) {
    (void)in_sizes; (void)n_in; (void)out_size; (void)ws_size;

    const int*   adj  = (const int*)  d_in[0];   // [2,E]
    const float* feat = (const float*)d_in[1];   // [N,128]
    const int*   et   = (const int*)  d_in[2];   // [E]
    const float* w    = (const float*)d_in[3];   // [R,128,64]
    const float* q    = (const float*)d_in[4];   // [64]
    const float* k    = (const float*)d_in[5];   // [64]
    const float* bias = (const float*)d_in[6];   // [64]
    float* out = (float*)d_out;                  // [N,64]

    // workspace layout (dwords)
    float* ws   = (float*)d_ws;
    float* h    = ws;                                   // R*N*OUTD
    float* qh   = h  + (size_t)RR * NN * OUTD;          // R*N
    float* kh   = qh + (size_t)RR * NN;                 // R*N
    float* exl  = kh + (size_t)RR * NN;                 // E
    float* den  = exl + (size_t)EE;                     // N
    unsigned* maxkey = (unsigned*)(den + NN);           // N
    unsigned* featbf = maxkey + NN;                     // N*IND/2 (16B-aligned)

    // 0. features -> packed bf16
    feat_to_bf16<<<(NN * (IND / 2) + 255) / 256, 256, 0, stream>>>(feat, featbf);

    // 1. per-relation GEMM via bf16 WMMA + fused q/k projections
    dim3 ggrid((NN + 127) / 128, RR);
    rgemm_wmma<<<ggrid, 256, 0, stream>>>(featbf, w, q, k, h, qh, kh);

    // 2. init
    init_buffers<<<(NN * OUTD + 255) / 256, 256, 0, stream>>>(out, den, maxkey);

    // 3. logits + segment max
    edge_logit_max<<<(EE + 255) / 256, 256, 0, stream>>>(adj, et, qh, kh, exl, maxkey);

    // 4. exp + segment sum
    edge_exp_sum<<<(EE + 255) / 256, 256, 0, stream>>>(adj, exl, maxkey, den);

    // 5. weighted scatter of messages (wave per edge)
    long long sthreads = (long long)EE * 32;
    edge_scatter<<<(unsigned)((sthreads + 255) / 256), 256, 0, stream>>>(adj, et, h, exl, den, out);

    // 6. bias + relu
    bias_relu<<<(NN * OUTD + 255) / 256, 256, 0, stream>>>(out, bias);
}
